// GraphConvolutionNetwork_80693845557544
// MI455X (gfx1250) — matrix operand order repaired and checked by reading, not compiled, not run on hardware
//
#include <hip/hip_runtime.h>

// GCN: 2x (GCNConv + ReLU). N=100000 nodes, E=1000000 edges, 128->64->64.
#define NNODES 100000
#define NEDGES 1000000
#define DIN    128
#define DHID   64
#define DOUTC  64

typedef __bf16 bf16_t;
typedef bf16_t   v16bf __attribute__((ext_vector_type(16)));
typedef float    v8f   __attribute__((ext_vector_type(8)));
typedef unsigned v8u   __attribute__((ext_vector_type(8)));

// Pack two f32 into two bf16 (truncation) with a single v_perm_b32:
// result[15:0] = lo[31:16], result[31:16] = hi[31:16].
__device__ __forceinline__ unsigned pack_bf16(float lo, float hi) {
  union { float f; unsigned u; } a, b;
  a.f = lo; b.f = hi;
  return __builtin_amdgcn_perm(b.u, a.u, 0x07060302u);
}

__device__ __forceinline__ void atomic_add_f32(float* p, float v) {
  __hip_atomic_fetch_add(p, v, __ATOMIC_RELAXED, __HIP_MEMORY_SCOPE_AGENT);
}

// ---------------- degree / normalization ----------------

__global__ void k_deg_init(float* __restrict__ deg, int n) {
  int i = blockIdx.x * blockDim.x + threadIdx.x;
  if (i < n) deg[i] = 1.0f;  // self-loop weight
}

__global__ void k_deg_scatter(const long long* __restrict__ col,
                              const float* __restrict__ w,
                              float* __restrict__ deg, int e) {
  int i = blockIdx.x * blockDim.x + threadIdx.x;
  if (i < e) atomic_add_f32(&deg[(int)col[i]], w[i]);
}

__global__ void k_deg_to_dinv(float* __restrict__ deg, int n) {
  int i = blockIdx.x * blockDim.x + threadIdx.x;
  if (i < n) {
    float d = deg[i];
    deg[i] = (d > 0.0f) ? rsqrtf(d) : 0.0f;
  }
}

// ---------------- weight pre-pack into WMMA B-fragment layout ----------------
// Wp[(ks*4 + t)*32 + lane] = 8 packed-bf16 dwords = lane's v16bf B fragment
// for k-step ks, 16-col tile t.  B packed in contiguous k order: lanes 0-15 hold
// K=0..15, lanes 16-31 hold K=16..31, VGPR v holds K pair {2v, 2v+1}.
template <int K>
__global__ void k_pack_w(const float* __restrict__ W, v8u* __restrict__ Wp) {
  const int total = (K / 32) * 4 * 32;
  int i = blockIdx.x * blockDim.x + threadIdx.x;
  if (i >= total) return;
  const int lane = i & 31;
  const int t    = (i >> 5) & 3;
  const int ks   = i >> 7;
  const int m    = lane & 15;
  const int half = lane >> 4;
  v8u bu;
#pragma unroll
  for (int v = 0; v < 8; ++v) {
    const int kbB = 16 * half + 2 * v;
    const float lo = W[(size_t)(ks * 32 + kbB)     * 64 + t * 16 + m];
    const float hi = W[(size_t)(ks * 32 + kbB + 1) * 64 + t * 16 + m];
    bu[v] = pack_bf16(lo, hi);
  }
  Wp[i] = bu;
}

// ---------------- WMMA GEMM: H[M,64] = A[M,K] @ W[K,64] ----------------
// One wave computes a full 16x64 output tile: A fragment built once per
// k-step (float2 loads + v_perm pack), B fragments are single 32B coalesced
// vector loads from the pre-packed weights; 4 WMMAs per k-step against 4
// accumulators. 8 row-tile waves per 256-thread block.
template <int K>
__global__ void k_gemm_wmma(const float* __restrict__ A,
                            const v8u* __restrict__ Wp,
                            float* __restrict__ H) {
  const int lane = threadIdx.x & 31;
  const int wave = threadIdx.x >> 5;
  const int row0 = (blockIdx.x * 8 + wave) * 16;
  if (row0 >= NNODES) return;
  const int m    = lane & 15;
  const int half = lane >> 4;

  v8f c0 = {}, c1 = {}, c2 = {}, c3 = {};
  const float* arow = A + (size_t)(row0 + m) * K;

  for (int k = 0; k < K; k += 32) {
    // A packed in contiguous k order: lanes 0-15 K in 0..15, lanes 16-31 K in 16..31 (matches packed B)
    v8u au;
#pragma unroll
    for (int v = 0; v < 8; ++v) {
      const int kbA = 16 * half + 2 * v;
      const float2 av = *(const float2*)(arow + k + kbA);
      au[v] = pack_bf16(av.x, av.y);
    }
    const v16bf a = __builtin_bit_cast(v16bf, au);
    const int ks = k >> 5;

#pragma unroll
    for (int t = 0; t < 4; ++t) {
      const v8u bu = Wp[((ks * 4 + t) << 5) + lane];
      const v16bf b = __builtin_bit_cast(v16bf, bu);
      v8f& c = (t == 0) ? c0 : (t == 1) ? c1 : (t == 2) ? c2 : c3;
      c = __builtin_amdgcn_wmma_f32_16x16x32_bf16(
          /*neg_a=*/false, a, /*neg_b=*/false, b,
          /*c_mod=*/(short)0, c, /*reuse_a=*/false, /*reuse_b=*/false);
    }
  }

  // C/D f32 layout: lane col n = lane&15, row = v + 8*(lane>>4)
#pragma unroll
  for (int t = 0; t < 4; ++t) {
    const v8f& c = (t == 0) ? c0 : (t == 1) ? c1 : (t == 2) ? c2 : c3;
#pragma unroll
    for (int v = 0; v < 8; ++v)
      H[(size_t)(row0 + v + 8 * half) * 64 + t * 16 + m] = c[v];
  }
}

// ---------------- aggregation ----------------

// out[n][j] = bias[j] + h[n][j] * dinv[n]^2   (bias + self-loop term)
__global__ void k_agg_init(const float* __restrict__ h,
                           const float* __restrict__ dinv,
                           const float* __restrict__ bias,
                           float* __restrict__ out, int total) {
  int i = blockIdx.x * blockDim.x + threadIdx.x;
  if (i < total) {
    int node = i >> 6;
    int j = i & 63;
    float di = dinv[node];
    out[i] = bias[j] + h[i] * di * di;
  }
}

// one wave32 per edge; lane handles 2 consecutive columns (float2 gather,
// two f32 atomics -> coalesced 256B bursts into L2-resident agg array)
__global__ void k_edge_scatter(const long long* __restrict__ row,
                               const long long* __restrict__ col,
                               const float* __restrict__ w,
                               const float* __restrict__ dinv,
                               const float* __restrict__ h,
                               float* __restrict__ out, int e) {
  int eid  = blockIdx.x * (blockDim.x >> 5) + (threadIdx.x >> 5);
  int lane = threadIdx.x & 31;
  if (eid < e) {
    long long r = row[eid];
    long long c = col[eid];
    float norm = dinv[r] * w[eid] * dinv[c];
    const float2 hv = *(const float2*)(h + (size_t)r * 64 + lane * 2);
    float* op = out + (size_t)c * 64 + lane * 2;
    atomic_add_f32(op,     hv.x * norm);
    atomic_add_f32(op + 1, hv.y * norm);
  }
}

__global__ void k_relu(float* __restrict__ x, int n) {
  int i = blockIdx.x * blockDim.x + threadIdx.x;
  if (i < n) x[i] = fmaxf(x[i], 0.0f);
}

// ---------------- launch ----------------

extern "C" void kernel_launch(void* const* d_in, const int* in_sizes, int n_in,
                              void* d_out, int out_size, void* d_ws, size_t ws_size,
                              hipStream_t stream) {
  const float*     x   = (const float*)d_in[0];
  const long long* ei  = (const long long*)d_in[1];  // int64 [2, E]
  const float*     ea  = (const float*)d_in[2];
  const float*     W1  = (const float*)d_in[3];
  const float*     b1  = (const float*)d_in[4];
  const float*     W2  = (const float*)d_in[5];
  const float*     b2  = (const float*)d_in[6];
  float*           out = (float*)d_out;

  const long long* erow = ei;            // edge_index[0]
  const long long* ecol = ei + NEDGES;   // edge_index[1]

  char* ws = (char*)d_ws;
  size_t off = 0;
  float* dinv = (float*)(ws + off); off += (((size_t)NNODES * 4 + 255) & ~(size_t)255);
  float* h    = (float*)(ws + off); off += (size_t)NNODES * 64 * 4;
  float* agg  = (float*)(ws + off); off += (size_t)NNODES * 64 * 4;
  v8u*   wp1  = (v8u*)(ws + off);   off += (size_t)(DIN  / 32) * 4 * 32 * 32;  // 16 KB
  v8u*   wp2  = (v8u*)(ws + off);   off += (size_t)(DHID / 32) * 4 * 32 * 32;  //  8 KB
  (void)ws_size; (void)in_sizes; (void)n_in; (void)out_size;

  const int TOTAL = NNODES * 64;
  const int ROWT  = NNODES / 16;               // 6250 row tiles (exact)
  const int GEMMB = (ROWT + 7) / 8;            // 8 row-tile waves per block
  dim3 b256(256);

  // --- pre-pack weights into WMMA B-fragment bf16 layout (tiny, once) ---
  k_pack_w<DIN ><<<2, b256, 0, stream>>>(W1, wp1);   // 512 fragments
  k_pack_w<DHID><<<1, b256, 0, stream>>>(W2, wp2);   // 256 fragments

  // --- symmetric normalization ---
  k_deg_init<<<(NNODES + 255) / 256, b256, 0, stream>>>(dinv, NNODES);
  k_deg_scatter<<<(NEDGES + 255) / 256, b256, 0, stream>>>(ecol, ea, dinv, NEDGES);
  k_deg_to_dinv<<<(NNODES + 255) / 256, b256, 0, stream>>>(dinv, NNODES);

  // --- layer 1: h = x @ W1 ; agg = b1 + selfloop + scatter ; relu ---
  k_gemm_wmma<DIN><<<GEMMB, b256, 0, stream>>>(x, wp1, h);
  k_agg_init<<<(TOTAL + 255) / 256, b256, 0, stream>>>(h, dinv, b1, agg, TOTAL);
  k_edge_scatter<<<(NEDGES + 7) / 8, b256, 0, stream>>>(erow, ecol, ea, dinv, h, agg, NEDGES);
  k_relu<<<(TOTAL + 255) / 256, b256, 0, stream>>>(agg, TOTAL);

  // --- layer 2: h = agg @ W2 ; out = b2 + selfloop + scatter ; relu ---
  k_gemm_wmma<DHID><<<GEMMB, b256, 0, stream>>>(agg, wp2, h);
  k_agg_init<<<(TOTAL + 255) / 256, b256, 0, stream>>>(h, dinv, b2, out, TOTAL);
  k_edge_scatter<<<(NEDGES + 7) / 8, b256, 0, stream>>>(erow, ecol, ea, dinv, h, out, NEDGES);
  k_relu<<<(TOTAL + 255) / 256, b256, 0, stream>>>(out, TOTAL);
}
